// DCConv2d_70050916597845
// MI455X (gfx1250) — matrix operand order repaired and checked by reading, not compiled, not run on hardware
//
#include <hip/hip_runtime.h>
#include <hip/hip_bf16.h>

typedef __bf16  v16bf __attribute__((ext_vector_type(16)));
typedef __bf16  v8bf  __attribute__((ext_vector_type(8)));
typedef __bf16  v4bf  __attribute__((ext_vector_type(4)));
typedef __bf16  v2bf  __attribute__((ext_vector_type(2)));
typedef float   v8f   __attribute__((ext_vector_type(8)));
typedef float   v4f   __attribute__((ext_vector_type(4)));

#define BDIM   128
#define CIN    32
#define COUT   32
#define HW     128
#define NUM_E  8
#define ROWS_PER_BLK 4
#define LDS_ROWS 6
#define LDS_W    132   // lds x = global x + 1; x in [-1, 130], padded to 132
#define LDS_C    40    // 32 channels padded to 40 (80B stride: 16B aligned, bank-conflict free)

// ---------------------------------------------------------------------------
// Kernel 1: dynamic weight generation, written directly in WMMA A-fragment
// order (16x32 bf16 A-matrix layout, ISA 7.12.2). One thread produces one
// VGPR-pair (2 bf16) => 4-byte coalesced stores.
// wfrag[b][oh][tap][lane][e], e in [0,16): e<8 -> c=8h+e, else c=16+8h+(e-8).
// ---------------------------------------------------------------------------
__global__ __launch_bounds__(256) void dyn_weight_frag(
    const float* __restrict__ se,          // (B, 8)
    const float* __restrict__ wbank,       // (O*C*9, 8) row-major
    unsigned short* __restrict__ wfrag_raw)
{
    __bf16* wfrag = (__bf16*)wfrag_raw;
    int idx  = blockIdx.x * 256 + threadIdx.x;   // 589,824 total
    int p    = idx & 7;                          // VGPR-pair index 0..7
    int lane = (idx >> 3) & 31;
    int rest = idx >> 8;
    int t    = rest % 9;  rest /= 9;             // tap
    int oh   = rest & 1;                         // out-channel half
    int b    = rest >> 1;                        // sample
    int h = lane >> 4, m = lane & 15;
    int o = oh * 16 + m;
    int c0 = (p < 4) ? (8 * h + 2 * p) : (16 + 8 * h + 2 * (p - 4));

    const float* seb = se + b * NUM_E;
    const float* w0 = wbank + (((o * CIN + c0    ) * 9) + t) * NUM_E;
    const float* w1 = wbank + (((o * CIN + c0 + 1) * 9) + t) * NUM_E;
    float s0 = 0.f, s1 = 0.f;
#pragma unroll
    for (int n = 0; n < NUM_E; ++n) {
        float sn = seb[n];
        s0 = fmaf(sn, w0[n], s0);
        s1 = fmaf(sn, w1[n], s1);
    }
    v2bf pv;
    pv[0] = (__bf16)s0;
    pv[1] = (__bf16)s1;
    *(v2bf*)(wfrag + (((((size_t)b * 2 + oh) * 9 + t) * 32 + lane) * 16 + 2 * p)) = pv;
}

// ---------------------------------------------------------------------------
// Kernel 2: implicit-GEMM conv via v_wmma_f32_16x16x32_bf16.
// Block = (sample b, 4 output rows x 128 cols). 8 waves:
//   wave -> (oh = out-channel half, pc = pixel-chunk phase), 8 16x16 tiles each.
// K-dim = 288 = 9 taps x 32 channels -> 9 WMMAs per tile.
// Input staged in LDS channel-contiguous bf16 [row][x][c]: each lane's
// B-fragment (channels 8h..8h+7 and 16+8h..16+8h+7) is two ds_load_b128.
// Staging uses aligned global_load_b128 (float4 along x) + register
// transpose + ds_store_b64; all task indexing is shift/mask only.
// ---------------------------------------------------------------------------
__global__ __launch_bounds__(256) void dyn_conv_wmma(
    const float* __restrict__ in,               // (B, C, H, W) f32
    const unsigned short* __restrict__ wfrag_raw,
    const float* __restrict__ bias,             // (O,)
    float* __restrict__ out)                    // (B, O, H, W) f32
{
    __shared__ __bf16 smem[LDS_ROWS * LDS_W * LDS_C];   // 63,360 B
    const __bf16* wfrag = (const __bf16*)wfrag_raw;

    const int b  = blockIdx.y;
    const int y0 = blockIdx.x * ROWS_PER_BLK;
    const float* inb = in + (size_t)b * CIN * HW * HW;

    // ---- Phase A: zero-fill halo columns lds_x in {0, 129, 130, 131}
    // 6 rows * 4 cols * 8 channel-quads = 192 tasks (disjoint from Phase B).
    if (threadIdx.x < 192) {
        int t    = threadIdx.x;
        int cq   = t & 7;
        int rest = t >> 3;            // 0..23
        int xi   = rest & 3;          // 0..3
        int row  = rest >> 2;         // 0..5
        int xl   = (xi == 0) ? 0 : (128 + xi);   // 0,129,130,131
        v4bf z = {};
        *(v4bf*)(smem + (row * LDS_W + xl) * LDS_C + cq * 4) = z;
    }

    // ---- Phase B: interior, lds_x in [1,129): aligned float4 loads along x.
    // Tasks: row(6) x xq(32 aligned x-quads) x cq(8 channel-quads) = 1536.
    for (int task = threadIdx.x; task < LDS_ROWS * 32 * 8; task += 256) {
        int cq  = task & 7;           // channel quad: c = cq*4 + i
        int xq  = (task >> 3) & 31;   // aligned x quad: x = xq*4 + j
        int row = task >> 8;          // 0..5
        int y   = y0 + row - 1;
        bool yok = (unsigned)y < (unsigned)HW;

        v4f q[4];
#pragma unroll
        for (int i = 0; i < 4; ++i) {
            v4f z = {0.f, 0.f, 0.f, 0.f};
            q[i] = z;
            if (yok)
                q[i] = *(const v4f*)(inb + (((cq * 4 + i) * HW) + y) * HW + xq * 4);
        }
#pragma unroll
        for (int j = 0; j < 4; ++j) {
            int xl = xq * 4 + j + 1;  // lds x = global x + 1
            v4bf v;
            v[0] = (__bf16)q[0][j];
            v[1] = (__bf16)q[1][j];
            v[2] = (__bf16)q[2][j];
            v[3] = (__bf16)q[3][j];
            *(v4bf*)(smem + (row * LDS_W + xl) * LDS_C + cq * 4) = v;  // ds_store_b64
        }
    }
    __syncthreads();

    const int lane = threadIdx.x & 31;
    const int wv   = threadIdx.x >> 5;
    const int oh   = wv & 1;       // out-channel half (o = oh*16 + m)
    const int pc   = wv >> 1;      // pixel-chunk phase 0..3
    const int h    = lane >> 4;
    const int m    = lane & 15;

    // A fragments (per-sample dynamic weights), pre-baked in fragment order.
    v16bf afrag[9];
    const __bf16* wfb = wfrag + ((((size_t)b * 2 + oh) * 9) * 32 + lane) * 16;
#pragma unroll
    for (int t = 0; t < 9; ++t)
        afrag[t] = *(const v16bf*)(wfb + t * 512);   // 32B contiguous per lane

    // bias for C/D fragment rows: VGPR v holds M = v + 8h
    float bias_v[8];
#pragma unroll
    for (int v = 0; v < 8; ++v)
        bias_v[v] = bias[oh * 16 + 8 * h + v];

    for (int j = 0; j < 8; ++j) {
        int ti = pc + 4 * j;       // tile id 0..31
        int r  = ti >> 3;          // output row in block (0..3)
        int cc = ti & 7;           // 16-pixel column chunk
        v8f acc = {0.f, 0.f, 0.f, 0.f, 0.f, 0.f, 0.f, 0.f};
#pragma unroll
        for (int t = 0; t < 9; ++t) {
            int ky = t / 3, kx = t % 3;
            int xl = cc * 16 + m + kx + 1;                  // +1: lds x = global x + 1, kx-1 tap offset
            const __bf16* bp = smem + ((r + ky) * LDS_W + xl - 1) * LDS_C + 8 * h;
            v8bf lo = *(const v8bf*)bp;                     // ds_load_b128
            v8bf hi = *(const v8bf*)(bp + 16);              // ds_load_b128
            v16bf bf = __builtin_shufflevector(lo, hi,
                0, 1, 2, 3, 4, 5, 6, 7, 8, 9, 10, 11, 12, 13, 14, 15);
            acc = __builtin_amdgcn_wmma_f32_16x16x32_bf16(
                false, afrag[t], false, bf, (short)0, acc, false, false);
        }
        // C/D layout: VGPR v -> o = oh*16 + 8h + v, n = lane&15
        float* op = out + (size_t)((b * COUT + oh * 16 + 8 * h) * HW + (y0 + r)) * HW
                        + cc * 16 + m;
#pragma unroll
        for (int v = 0; v < 8; ++v)
            op[(size_t)v * HW * HW] = acc[v] + bias_v[v];
    }
}

// ---------------------------------------------------------------------------
extern "C" void kernel_launch(void* const* d_in, const int* in_sizes, int n_in,
                              void* d_out, int out_size, void* d_ws, size_t ws_size,
                              hipStream_t stream) {
    (void)in_sizes; (void)n_in; (void)out_size; (void)ws_size;
    const float* inputs = (const float*)d_in[0];   // (128,32,128,128)
    const float* se     = (const float*)d_in[1];   // (128,8)
    const float* wbank  = (const float*)d_in[2];   // (9216,8)
    const float* bias   = (const float*)d_in[3];   // (32,)
    float* out = (float*)d_out;
    unsigned short* wfrag = (unsigned short*)d_ws; // 2.36 MB bf16 fragments

    // 128*2*9*32*8 pair-tasks / 256 = 2304 blocks
    dyn_weight_frag<<<2304, 256, 0, stream>>>(se, wbank, wfrag);
    // (H/4) y-tiles x B samples
    dyn_conv_wmma<<<dim3(HW / ROWS_PER_BLK, BDIM), 256, 0, stream>>>(inputs, wfrag, bias, out);
}
